// LinearCDE_56195352101066
// MI455X (gfx1250) — matrix operand order repaired and checked by reading, not compiled, not run on hardware
//
#include <hip/hip_runtime.h>

typedef __attribute__((ext_vector_type(16))) _Float16 v16h;
typedef __attribute__((ext_vector_type(8)))  float    v8f;

constexpr int Bsz = 32;
constexpr int L   = 2048;
constexpr int D   = 127;
constexpr int H   = 256;
constexpr int OMG = 128;          // omega = D + 1
constexpr int KTOT = OMG * H;     // 32768  (K of the skinny GEMM)
constexpr int KC   = KTOT / 32;   // 1024 K-chunks of 32
constexpr float DT = 1.0f / (float)L;

// ---------------------------------------------------------------------------
// Convert vf_A (H x OMG x H, fp32) -> f16, pre-swizzled into the exact WMMA
// A-fragment layout (16-bit A 16x32, ISA 7.12.2), so a wave's fragment for
// (mi, kc) is 32 contiguous bytes per lane:
//   A16s[(((mi*1024 + kc)*32 + lane)*16 + j]
// lane 0-15 : M = mi*16 + lane,  VGPR v: K = (v<4?0:16) + (v&3)*2 + pos
// lane16-31 : same M,            K += 8
// ---------------------------------------------------------------------------
__global__ void __launch_bounds__(256) cvtA_kernel(const float* __restrict__ vfA,
                                                   _Float16* __restrict__ A16s) {
    int tid  = blockIdx.x * 256 + threadIdx.x;     // 0 .. 2^23-1
    int j    = tid & 15;
    int lane = (tid >> 4) & 31;
    int kc   = (tid >> 9) & 1023;
    int mi   = tid >> 19;
    int m    = mi * 16 + (lane & 15);
    int h2   = lane >> 4;
    int v    = j >> 1;
    int pos  = j & 1;
    int kl   = ((v & 4) ? 16 : 0) + h2 * 8 + (v & 3) * 2 + pos;
    int Kg   = kc * 32 + kl;                       // flat K = o*H + kk
    int o    = Kg >> 8;                            // / H
    int kk   = Kg & 255;                           // % H
    A16s[tid] = (_Float16)vfA[((size_t)m * OMG + o) * H + kk];
}

// ---------------------------------------------------------------------------
// y0 = X[:,0,:] @ init_matrix^T + init_bias        (B x H, fp32)
// ---------------------------------------------------------------------------
__global__ void __launch_bounds__(256) y0_kernel(const float* __restrict__ X,
                                                 const float* __restrict__ M,
                                                 const float* __restrict__ bias,
                                                 float* __restrict__ y) {
    int tid = blockIdx.x * 256 + threadIdx.x;      // 0 .. B*H-1
    int b = tid >> 8, h = tid & 255;
    float acc = bias[h];
    const float* xr = X + (size_t)b * L * D;       // row l = 0
    const float* mr = M + (size_t)h * D;
    for (int d = 0; d < D; ++d) acc = fmaf(xr[d], mr[d], acc);
    y[b * H + h] = acc;
}

// ---------------------------------------------------------------------------
// One Euler step, fully fused:
//   D(256x32) = A_flat(256x32768) @ P(32768x32),  P[o*H+k,b] = c[b,o]*y[b,k]
//   y_out = y_in + dt * (D + c @ vf_B^T)
// grid = 32 WGs (16 M-tiles x 2 N-tiles), 256 threads = 8 wave32 waves.
// Waves split K contiguously (128 chunks each), LDS tree-reduce, wave 0
// applies bias term + update. Deterministic (no atomics).
// ---------------------------------------------------------------------------
__global__ void __launch_bounds__(256) step_kernel(const float* __restrict__ Xg,
                                                   const float* __restrict__ vfB,
                                                   const _Float16* __restrict__ A16s,
                                                   const float* __restrict__ y_in,
                                                   float* __restrict__ y_out,
                                                   int l) {
    __shared__ __align__(32) _Float16 y_sm[H][16];    // [kk][b]  8 KB
    __shared__ __align__(32) _Float16 c_sm[OMG][16];  // [o][b]   4 KB
    __shared__ float c32[16][132];                    // [b][o]   padded, 8.25 KB
    __shared__ float bt[16][17];                      // bias-term tile
    __shared__ float red[8][32][8];                   // cross-wave reduce, 8 KB

    const int tid  = threadIdx.x;
    const int lane = tid & 31;
    const int w    = tid >> 5;
    const int wg   = blockIdx.x;
    const int mi   = wg & 15;          // M tile (h0 = mi*16)
    const int ni   = wg >> 4;          // N tile (b0 = ni*16)
    const int b0   = ni * 16, h0 = mi * 16;
    const float t  = (float)l * (1.0f / (float)(L - 1));

    // ---- stage y (fp32 -> f16) and control c = [t, X_l] ----
    for (int i = tid; i < H * 16; i += 256) {
        int kk = i >> 4, b = i & 15;
        y_sm[kk][b] = (_Float16)y_in[(b0 + b) * H + kk];
    }
    for (int i = tid; i < OMG * 16; i += 256) {
        int o = i >> 4, b = i & 15;
        float v = (o == 0) ? t : Xg[(((size_t)(b0 + b)) * L + l) * D + (o - 1)];
        c_sm[o][b] = (_Float16)v;
        c32[b][o]  = v;
    }
    __syncthreads();

    // ---- bias term tile: bt[b][hl] = sum_o c[b,o] * vf_B[h0+hl, o]
    // one 128-dot per lane (256 lanes = 16x16 tile), overlaps the GEMM setup.
    {
        int b = tid & 15, hl = tid >> 4;
        const float* vb = vfB + (size_t)(h0 + hl) * OMG;
        float a = 0.0f;
        for (int o = 0; o < OMG; ++o) a = fmaf(c32[b][o], vb[o], a);
        bt[b][hl] = a;
    }

    // ---- main K loop: wave w owns K-chunks [w*128, (w+1)*128) ----
    v8f acc = {};
    const _Float16* Abase = A16s + (size_t)mi * KC * 32 * 16;
    for (int kc = w * 128; kc < (w + 1) * 128; ++kc) {
        // A fragment: contiguous 32 B per lane (pre-swizzled)
        v16h a = *(const v16h*)(Abase + ((size_t)kc * 32 + lane) * 16);
        // B fragment built on the fly: lane <-> N col, 16 halfwords <-> K rows (table order)
        int o  = kc >> 3;                 // K block of 256 -> omega index
        int n  = lane & 15;
        int kb = ((kc & 7) << 5) + (lane >> 4) * 8;
        _Float16 cn = c_sm[o][n];
        v16h bf;
        for (int s = 0; s < 16; ++s) {
            int kk = kb + (s < 8 ? s : 8 + s);  // k index within H
            bf[s] = cn * y_sm[kk][n];
        }
        acc = __builtin_amdgcn_wmma_f32_16x16x32_f16(
                  false, a, false, bf, (short)0, acc, false, false);
    }

    // ---- cross-wave reduce + fused update ----
    for (int j = 0; j < 8; ++j) red[w][lane][j] = acc[j];
    __syncthreads();
    if (w == 0) {
        v8f s = {};
        for (int ww = 0; ww < 8; ++ww)
            for (int j = 0; j < 8; ++j) s[j] += red[ww][lane][j];
        // D layout: VGPR j -> M = j + 8*(lane>>4), N = lane&15
        int b  = lane & 15;
        int hh = (lane >> 4) * 8;
        for (int j = 0; j < 8; ++j) {
            int hl = hh + j;
            int gb = b0 + b, gh = h0 + hl;
            float yo = y_in[gb * H + gh];
            y_out[gb * H + gh] = fmaf(DT, s[j] + bt[b][hl], yo);
        }
    }
}

// ---------------------------------------------------------------------------
extern "C" void kernel_launch(void* const* d_in, const int* in_sizes, int n_in,
                              void* d_out, int out_size, void* d_ws, size_t ws_size,
                              hipStream_t stream) {
    const float* X     = (const float*)d_in[0];  // (B, L, D)
    const float* initM = (const float*)d_in[1];  // (H, D)
    const float* initb = (const float*)d_in[2];  // (H,)
    const float* vfA   = (const float*)d_in[3];  // (H, OMG, H)
    const float* vfB   = (const float*)d_in[4];  // (H, OMG)
    float* out = (float*)d_out;                  // (B, H)

    char* ws = (char*)d_ws;
    _Float16* A16s = (_Float16*)ws;                                   // 16 MiB
    float* buf0 = (float*)(ws + (size_t)16 * 1024 * 1024);            // B*H fp32
    float* buf1 = buf0 + Bsz * H;

    // one-time: f16-swizzle vf_A (L2-resident afterwards), y0 init
    cvtA_kernel<<<(16 * KC * 32 * 16) / 256, 256, 0, stream>>>(vfA, A16s);
    y0_kernel<<<(Bsz * H) / 256, 256, 0, stream>>>(X, initM, initb, buf0);

    // sequential Euler scan: steps use control at time index l = 1 .. L-1
    for (int l = 1; l < L; ++l) {
        const float* yin = (l & 1) ? buf0 : buf1;
        float* yout = (l == L - 1) ? out : ((l & 1) ? buf1 : buf0);
        step_kernel<<<32, 256, 0, stream>>>(X, vfB, A16s, yin, yout, l);
    }
}
